// FCB_41609643163755
// MI455X (gfx1250) — compile-verified, matrix operand order checked by reading, not run on hardware
//
#include <hip/hip_runtime.h>

// Spectral conv y = irfft2(rfft2(x) * w) as FFT-by-GEMM on v_wmma_f32_16x16x32_bf16.
// All operands pre-packed in ISA WMMA fragment layout (hi/lo split-bf16, 3-product
// per logical fp32 GEMM). Steady-state loops: b128 loads + wmma only.

#define HH   256
#define WFQ  129           // rfft bins
#define NIMG 1024          // B*C
#define NC   64

// fragment geometry: A = 16x32 (MxK), B = 32x16 (KxN), 512 bf16 halves / fragment
// per-matrix fragment counts:
//   Fh/Gh (A, 256x256): [mt 16][kc 8]    = 128 frags
//   Fw    (B, 256x160): [kc 8][nt 10]    = 80 frags
//   Gw    (B, 160x256): [kc 5][nt 16]    = 80 frags
//   T     (B, 256x160 per img): [kc 8][jt 10] = 80 frags
//   Z     (A, 256x160 per img): [mt 16][kc 5] = 80 frags
#define FRAG  512
#define TZH   40960        // 80 frags * 512 halves, per component per image

typedef __attribute__((ext_vector_type(16))) __bf16 v16bf;
typedef __attribute__((ext_vector_type(8)))  float  v8f;

// K index inside a 32-wide fragment for element slot s (0..15) and lane group g (0/1).
__device__ __forceinline__ int kpat(int s, int g) {
    int rp = s >> 1;
    return ((rp < 4) ? 0 : 16) + g * 8 + (rp & 3) * 2 + (s & 1);
}
// inverse: K index -> (lane group, slot)
__device__ __forceinline__ void kinv(int kk, int& g, int& s) {
    g = (kk >> 3) & 1;
    s = (((kk >> 4) * 4 + ((kk >> 1) & 3)) << 1) | (kk & 1);
}
__device__ __forceinline__ __bf16 bhi(float v) { return (__bf16)v; }
__device__ __forceinline__ __bf16 blo(float v) { __bf16 h = (__bf16)v; return (__bf16)(v - (float)h); }

// packed fragment load: lane reads its 16 contiguous halves (2 x b128)
__device__ __forceinline__ v16bf ldfrag(const __bf16* base, int frag) {
    return *(const v16bf*)(base + (((size_t)frag * 32 + (threadIdx.x & 31)) << 4));
}

// split-bf16 3-product accumulate: acc += Ah*Bh + Ah*Bl + Al*Bh
__device__ __forceinline__ v8f wmma3(v16bf ah, v16bf al, v16bf bh, v16bf bl, v8f acc) {
    acc = __builtin_amdgcn_wmma_f32_16x16x32_bf16(false, ah, false, bh, (short)0, acc, false, false);
    acc = __builtin_amdgcn_wmma_f32_16x16x32_bf16(false, ah, false, bl, (short)0, acc, false, false);
    acc = __builtin_amdgcn_wmma_f32_16x16x32_bf16(false, al, false, bh, (short)0, acc, false, false);
    return acc;
}

// ---------------- packed DFT table builders (run once per launch) ----------------
__global__ void k_tabA(__bf16* FhrH, __bf16* FhrL, __bf16* FhiH, __bf16* FhiL,
                       __bf16* GhrH, __bf16* GhrL, __bf16* GhiH, __bf16* GhiL) {
    int tid = blockIdx.x * blockDim.x + threadIdx.x;      // 16*8*32 = 4096
    if (tid >= 4096) return;
    int lane = tid & 31, f = tid >> 5;
    int kc = f & 7, mt = f >> 3;
    int m = mt * 16 + (lane & 15);
    int g = lane >> 4;
    const float step = 6.2831853071795864769f / 256.0f;
    int base = tid << 4;
#pragma unroll
    for (int s = 0; s < 16; ++s) {
        int kk = kc * 32 + kpat(s, g);
        float th = (float)((m * kk) & 255) * step;
        float sn, cs; __sincosf(th, &sn, &cs);
        FhrH[base + s] = bhi(cs);  FhrL[base + s] = blo(cs);
        FhiH[base + s] = bhi(-sn); FhiL[base + s] = blo(-sn);
        GhrH[base + s] = bhi(cs);  GhrL[base + s] = blo(cs);
        GhiH[base + s] = bhi(sn);  GhiL[base + s] = blo(sn);
    }
}

__global__ void k_tabFw(__bf16* FwrH, __bf16* FwrL, __bf16* FwiH, __bf16* FwiL) {
    int tid = blockIdx.x * blockDim.x + threadIdx.x;      // 8*10*32 = 2560
    if (tid >= 2560) return;
    int lane = tid & 31, f = tid >> 5;                    // f = kc*10 + nt
    int kc = f / 10;
    int nt = f - kc * 10;
    int n = nt * 16 + (lane & 15);                        // freq bin (0..159)
    int g = lane >> 4;
    const float step = 6.2831853071795864769f / 256.0f;
    int base = tid << 4;
#pragma unroll
    for (int s = 0; s < 16; ++s) {
        int kk = kc * 32 + kpat(s, g);                    // spatial w (0..255)
        float vr = 0.f, vi = 0.f;
        if (n < WFQ) {
            float th = (float)((kk * n) & 255) * step;
            float sn, cs; __sincosf(th, &sn, &cs);
            vr = cs; vi = -sn;
        }
        FwrH[base + s] = bhi(vr); FwrL[base + s] = blo(vr);
        FwiH[base + s] = bhi(vi); FwiL[base + s] = blo(vi);
    }
}

__global__ void k_tabGw(__bf16* GwcH, __bf16* GwcL, __bf16* GwsH, __bf16* GwsL) {
    int tid = blockIdx.x * blockDim.x + threadIdx.x;      // 5*16*32 = 2560
    if (tid >= 2560) return;
    int lane = tid & 31, f = tid >> 5;                    // f = kc*16 + nt
    int kc = f >> 4, nt = f & 15;
    int n = nt * 16 + (lane & 15);                        // spatial w (0..255)
    int g = lane >> 4;
    const float step = 6.2831853071795864769f / 256.0f;
    int base = tid << 4;
#pragma unroll
    for (int s = 0; s < 16; ++s) {
        int kk = kc * 32 + kpat(s, g);                    // freq bin (0..159)
        float gc = 0.f, gs = 0.f;
        if (kk < WFQ) {
            float ck = (kk == 0 || kk == 128) ? 1.0f : 2.0f;
            float sc = ck * (1.0f / 65536.0f);            // hermitian weight * 1/(H*W)
            float th = (float)((kk * n) & 255) * step;
            float sn, cs; __sincosf(th, &sn, &cs);
            gc = cs * sc;                                 // Re(Z e^{+i th}) = Zr*cos - Zi*sin
            gs = -sn * sc;
        }
        GwcH[base + s] = bhi(gc); GwcL[base + s] = blo(gc);
        GwsH[base + s] = bhi(gs); GwsL[base + s] = blo(gs);
    }
}

// ---------------- Stage 1: row rFFT  T = x @ Fw ----------------
// grid(8 strips, nimg), block 320 = 10 waves (wave = 16-col freq tile).
// x strip is split+A-packed into LDS once per block; T written scatter-packed (B layout).
__global__ void __launch_bounds__(320) k_rowfft(const float* __restrict__ x,
        const __bf16* __restrict__ FwrH, const __bf16* __restrict__ FwrL,
        const __bf16* __restrict__ FwiH, const __bf16* __restrict__ FwiL,
        __bf16* __restrict__ TrH, __bf16* __restrict__ TrL,
        __bf16* __restrict__ TiH, __bf16* __restrict__ TiL, int img0) {
    __shared__ alignas(32) __bf16 Apack[2][8][2][32][16];   // [rt][kc][h/l][lane][slot] 32KB
    const int tid = threadIdx.x;
    const int rb = blockIdx.x;                              // 32-row strip
    const int iml = blockIdx.y;
    const float* xi = x + ((size_t)(img0 + iml) * HH + rb * 32) * HH;

    for (int u = tid; u < 512; u += 320) {                  // pack x strip (once per block)
        int lane = u & 31, f = u >> 5;                      // f = rt*8 + kc
        int kc = f & 7, rt = f >> 3;
        const float* p = xi + (rt * 16 + (lane & 15)) * HH + kc * 32 + (lane >> 4) * 8;
        __bf16* dh = &Apack[rt][kc][0][lane][0];
        __bf16* dl = &Apack[rt][kc][1][lane][0];
#pragma unroll
        for (int r = 0; r < 8; ++r) {
            int kk = ((r < 4) ? 0 : 16) + (r & 3) * 2;
            float f0 = p[kk], f1 = p[kk + 1];
            dh[2 * r]     = bhi(f0); dl[2 * r]     = blo(f0);
            dh[2 * r + 1] = bhi(f1); dl[2 * r + 1] = blo(f1);
        }
    }
    __syncthreads();

    const int wv = tid >> 5, lane = tid & 31;
    v8f aR0 = {}, aI0 = {}, aR1 = {}, aI1 = {};
    for (int kc = 0; kc < 8; ++kc) {
        v16bf a0h = *(const v16bf*)&Apack[0][kc][0][lane][0];
        v16bf a0l = *(const v16bf*)&Apack[0][kc][1][lane][0];
        v16bf a1h = *(const v16bf*)&Apack[1][kc][0][lane][0];
        v16bf a1l = *(const v16bf*)&Apack[1][kc][1][lane][0];
        int fb = kc * 10 + wv;
        v16bf brh = ldfrag(FwrH, fb), brl = ldfrag(FwrL, fb);
        v16bf bih = ldfrag(FwiH, fb), bil = ldfrag(FwiL, fb);
        aR0 = wmma3(a0h, a0l, brh, brl, aR0);
        aI0 = wmma3(a0h, a0l, bih, bil, aI0);
        aR1 = wmma3(a1h, a1l, brh, brl, aR1);
        aI1 = wmma3(a1h, a1l, bih, bil, aI1);
    }
    // scatter-store T as packed B fragment [kc=rb][jt=wv]
    const size_t toff = (size_t)iml * TZH;
    const size_t fragBase = ((size_t)(rb * 10 + wv) * 32) << 4;
#pragma unroll
    for (int rt = 0; rt < 2; ++rt) {
        v8f aR = rt ? aR1 : aR0;
        v8f aI = rt ? aI1 : aI0;
#pragma unroll
        for (int r = 0; r < 8; ++r) {
            int kk = rt * 16 + ((lane >> 4) << 3) + r;      // K row within fragment
            int gg, ss; kinv(kk, gg, ss);
            size_t idx = toff + fragBase + (((lane & 15) + 16 * gg) << 4) + ss;
            TrH[idx] = bhi(aR[r]); TrL[idx] = blo(aR[r]);
            TiH[idx] = bhi(aI[r]); TiL[idx] = blo(aI[r]);
        }
    }
}

// ---------------- Stage 2 (fused): U = Fh@T ; V = U.*w ; Z = Gh@V ----------------
// grid(5 col-pairs, nimg), block 512 = 16 waves (wave = 16-row tile, 32 cols).
// V lives only as a packed B-fragment LDS image; Z scatter-stored in A layout.
__global__ void __launch_bounds__(512) k_colfilt(
        const __bf16* __restrict__ TrH, const __bf16* __restrict__ TrL,
        const __bf16* __restrict__ TiH, const __bf16* __restrict__ TiL,
        const __bf16* __restrict__ FhrH, const __bf16* __restrict__ FhrL,
        const __bf16* __restrict__ FhiH, const __bf16* __restrict__ FhiL,
        const __bf16* __restrict__ GhrH, const __bf16* __restrict__ GhrL,
        const __bf16* __restrict__ GhiH, const __bf16* __restrict__ GhiL,
        const float* __restrict__ wr, const float* __restrict__ wi,
        __bf16* __restrict__ ZrH, __bf16* __restrict__ ZrL,
        __bf16* __restrict__ ZiH, __bf16* __restrict__ ZiL, int img0) {
    __shared__ alignas(32) __bf16 Vp[8][2][4][32][16];      // [kc][lt][rh,rl,ih,il][lane][slot] 64KB
    const int tid = threadIdx.x;
    const int wv = tid >> 5, lane = tid & 31;
    const int m0 = wv * 16;
    const int jp = blockIdx.x;                              // 32-col block
    const int iml = blockIdx.y;
    const int ch = (img0 + iml) & (NC - 1);                 // img = b*C + c
    const size_t toff = (size_t)iml * TZH;

    // Phase A: forward column FFT (complex), 2 col tiles per wave
    v8f RR[2] = {}, II[2] = {}, RI[2] = {}, IR[2] = {};
    for (int kc = 0; kc < 8; ++kc) {
        int fa = wv * 8 + kc;
        v16bf arh = ldfrag(FhrH, fa), arl = ldfrag(FhrL, fa);
        v16bf aih = ldfrag(FhiH, fa), ail = ldfrag(FhiL, fa);
#pragma unroll
        for (int lt = 0; lt < 2; ++lt) {
            int fb = kc * 10 + jp * 2 + lt;
            v16bf brh = ldfrag(TrH + toff, fb), brl = ldfrag(TrL + toff, fb);
            v16bf bih = ldfrag(TiH + toff, fb), bil = ldfrag(TiL + toff, fb);
            RR[lt] = wmma3(arh, arl, brh, brl, RR[lt]);
            II[lt] = wmma3(aih, ail, bih, bil, II[lt]);
            RI[lt] = wmma3(arh, arl, bih, bil, RI[lt]);
            IR[lt] = wmma3(aih, ail, brh, brl, IR[lt]);
        }
    }
    // pointwise spectral filter; scatter V straight into packed B-fragment LDS
    {
        const float* wrc = wr + (size_t)ch * HH * WFQ;
        const float* wic = wi + (size_t)ch * HH * WFQ;
        const int mb = m0 + ((lane >> 4) << 3);
#pragma unroll
        for (int lt = 0; lt < 2; ++lt) {
            int n = jp * 32 + lt * 16 + (lane & 15);        // freq bin
#pragma unroll
            for (int r = 0; r < 8; ++r) {
                int h = mb + r;
                float ur = RR[lt][r] - II[lt][r];
                float ui = RI[lt][r] + IR[lt][r];
                float fr = 0.f, fi = 0.f;
                if (n < WFQ) { fr = wrc[h * WFQ + n]; fi = wic[h * WFQ + n]; }
                float vr = ur * fr - ui * fi;
                float vi = ur * fi + ui * fr;
                int gg, ss; kinv(h & 31, gg, ss);
                int tl = (lane & 15) + 16 * gg;
                int kcf = h >> 5;
                Vp[kcf][lt][0][tl][ss] = bhi(vr);
                Vp[kcf][lt][1][tl][ss] = blo(vr);
                Vp[kcf][lt][2][tl][ss] = bhi(vi);
                Vp[kcf][lt][3][tl][ss] = blo(vi);
            }
        }
    }
    __syncthreads();

    // Phase B: inverse column FFT  Z = Gh @ V  (B operand from packed LDS)
    v8f zRR[2] = {}, zII[2] = {}, zRI[2] = {}, zIR[2] = {};
    for (int kc = 0; kc < 8; ++kc) {
        int fa = wv * 8 + kc;
        v16bf arh = ldfrag(GhrH, fa), arl = ldfrag(GhrL, fa);
        v16bf aih = ldfrag(GhiH, fa), ail = ldfrag(GhiL, fa);
#pragma unroll
        for (int lt = 0; lt < 2; ++lt) {
            v16bf brh = *(const v16bf*)&Vp[kc][lt][0][lane][0];
            v16bf brl = *(const v16bf*)&Vp[kc][lt][1][lane][0];
            v16bf bih = *(const v16bf*)&Vp[kc][lt][2][lane][0];
            v16bf bil = *(const v16bf*)&Vp[kc][lt][3][lane][0];
            zRR[lt] = wmma3(arh, arl, brh, brl, zRR[lt]);
            zII[lt] = wmma3(aih, ail, bih, bil, zII[lt]);
            zRI[lt] = wmma3(arh, arl, bih, bil, zRI[lt]);
            zIR[lt] = wmma3(aih, ail, brh, brl, zIR[lt]);
        }
    }
    // scatter-store Z as packed A fragment [mt=wv][kc=jp]
    const size_t zoff = (size_t)iml * TZH;
    const size_t fragBase = ((size_t)(wv * 5 + jp) * 32) << 4;
    const int mloc0 = (lane >> 4) << 3;
#pragma unroll
    for (int lt = 0; lt < 2; ++lt) {
        int c = lt * 16 + (lane & 15);                      // K col within fragment
        int gg, ss; kinv(c, gg, ss);
#pragma unroll
        for (int r = 0; r < 8; ++r) {
            float zr = zRR[lt][r] - zII[lt][r];
            float zi = zRI[lt][r] + zIR[lt][r];
            size_t idx = zoff + fragBase + (((mloc0 + r) + 16 * gg) << 4) + ss;
            ZrH[idx] = bhi(zr); ZrL[idx] = blo(zr);
            ZiH[idx] = bhi(zi); ZiL[idx] = blo(zi);
        }
    }
}

// ---------------- Stage 3: inverse row rFFT (real)  y = Zr@Gwc + Zi@Gws ----------------
// grid(16 rowtiles, nimg), block 512 = 16 waves (wave = 16-col output tile). Pure packed loads.
__global__ void __launch_bounds__(512) k_irow(
        const __bf16* __restrict__ ZrH, const __bf16* __restrict__ ZrL,
        const __bf16* __restrict__ ZiH, const __bf16* __restrict__ ZiL,
        const __bf16* __restrict__ GwcH, const __bf16* __restrict__ GwcL,
        const __bf16* __restrict__ GwsH, const __bf16* __restrict__ GwsL,
        float* __restrict__ y, int img0) {
    const int tid = threadIdx.x;
    const int nt = tid >> 5, lane = tid & 31;
    const int mt = blockIdx.x;
    const int iml = blockIdx.y;
    const size_t zoff = (size_t)iml * TZH;
    v8f acc = {};
    for (int kc = 0; kc < 5; ++kc) {
        int fa = mt * 5 + kc;
        v16bf zrh = ldfrag(ZrH + zoff, fa), zrl = ldfrag(ZrL + zoff, fa);
        v16bf zih = ldfrag(ZiH + zoff, fa), zil = ldfrag(ZiL + zoff, fa);
        int fb = kc * 16 + nt;
        v16bf bch = ldfrag(GwcH, fb), bcl = ldfrag(GwcL, fb);
        v16bf bsh = ldfrag(GwsH, fb), bsl = ldfrag(GwsL, fb);
        acc = wmma3(zrh, zrl, bch, bcl, acc);
        acc = wmma3(zih, zil, bsh, bsl, acc);
    }
    float* yo = y + (size_t)(img0 + iml) * HH * HH;
    const int n = nt * 16 + (lane & 15);
    const int mb = mt * 16 + ((lane >> 4) << 3);
#pragma unroll
    for (int r = 0; r < 8; ++r) yo[(size_t)(mb + r) * HH + n] = acc[r];
}

extern "C" void kernel_launch(void* const* d_in, const int* in_sizes, int n_in,
                              void* d_out, int out_size, void* d_ws, size_t ws_size,
                              hipStream_t stream) {
    (void)in_sizes; (void)n_in; (void)out_size;
    const float* x      = (const float*)d_in[0];
    const float* w_real = (const float*)d_in[1];
    const float* w_imag = (const float*)d_in[2];
    float* y = (float*)d_out;

    __bf16* p = (__bf16*)d_ws;
    size_t o = 0;
    auto A = [&](size_t n) { __bf16* r = p + o; o += n; return r; };

    // packed tables (halves)
    __bf16 *FhrH = A(65536), *FhrL = A(65536), *FhiH = A(65536), *FhiL = A(65536);
    __bf16 *GhrH = A(65536), *GhrL = A(65536), *GhiH = A(65536), *GhiL = A(65536);
    __bf16 *FwrH = A(40960), *FwrL = A(40960), *FwiH = A(40960), *FwiL = A(40960);
    __bf16 *GwcH = A(40960), *GwcL = A(40960), *GwsH = A(40960), *GwsL = A(40960);
    const size_t tabHalves = o;

    // chunk sizing: 8 packed planes of TZH halves per in-flight image (640 KB/img),
    // sized so the T/Z working set stays L2-resident (192 MB).
    size_t availH = (ws_size / 2 > tabHalves) ? (ws_size / 2 - tabHalves) : 0;
    int chunk = (int)(availH / (8ull * TZH));
    if (chunk > 128) chunk = 128;
    if (chunk < 1)   chunk = 1;

    __bf16 *TrH = A((size_t)chunk * TZH), *TrL = A((size_t)chunk * TZH);
    __bf16 *TiH = A((size_t)chunk * TZH), *TiL = A((size_t)chunk * TZH);
    __bf16 *ZrH = A((size_t)chunk * TZH), *ZrL = A((size_t)chunk * TZH);
    __bf16 *ZiH = A((size_t)chunk * TZH), *ZiL = A((size_t)chunk * TZH);

    k_tabA <<<16, 256, 0, stream>>>(FhrH, FhrL, FhiH, FhiL, GhrH, GhrL, GhiH, GhiL);
    k_tabFw<<<10, 256, 0, stream>>>(FwrH, FwrL, FwiH, FwiL);
    k_tabGw<<<10, 256, 0, stream>>>(GwcH, GwcL, GwsH, GwsL);

    for (int img0 = 0; img0 < NIMG; img0 += chunk) {
        int n = NIMG - img0; if (n > chunk) n = chunk;
        k_rowfft <<<dim3(8, n),  320, 0, stream>>>(x, FwrH, FwrL, FwiH, FwiL,
                                                   TrH, TrL, TiH, TiL, img0);
        k_colfilt<<<dim3(5, n),  512, 0, stream>>>(TrH, TrL, TiH, TiL,
                                                   FhrH, FhrL, FhiH, FhiL,
                                                   GhrH, GhrL, GhiH, GhiL,
                                                   w_real, w_imag,
                                                   ZrH, ZrL, ZiH, ZiL, img0);
        k_irow   <<<dim3(16, n), 512, 0, stream>>>(ZrH, ZrL, ZiH, ZiL,
                                                   GwcH, GwcL, GwsH, GwsL, y, img0);
    }
}
